// EMAVectorQuantizer_39170101740127
// MI455X (gfx1250) — matrix operand order repaired and checked by reading, not compiled, not run on hardware
//
#include <hip/hip_runtime.h>
#include <hip/hip_bf16.h>

typedef __attribute__((ext_vector_type(16))) __bf16 v16bf;
typedef __attribute__((ext_vector_type(8)))  float  v8f;

#define DIM      256
#define NTOK     16384     // 16 * 32 * 32
#define NEMB     16384
#define FIXLEN   30.0f
#define ZQ_ELEMS 4194304   // 16*256*32*32

// LDS staging for the argmin GEMM
#define STAGE_ROWS 32                  // embedding rows per stage
#define ROW_US     264                 // 256 data ushorts + 8 pad (528 B): bank-conflict-free
#define NSTAGES    (NEMB / STAGE_ROWS) // 512

// round-to-nearest-even fp32 -> bf16 bits
__device__ __forceinline__ unsigned short f2bf(float f) {
    unsigned int u = __float_as_uint(f);
    unsigned int r = u + 0x7FFFu + ((u >> 16) & 1u);
    return (unsigned short)(r >> 16);
}

union V16U { v16bf v; uint4 u[2]; };

// ---------------------------------------------------------------------------
// 1) sum of per-token channel L2 norms.  z is [b=16][c=256][hw=1024].
// ---------------------------------------------------------------------------
__global__ void k_norm(const float* __restrict__ z, float* __restrict__ sums) {
    int t  = blockIdx.x * 256 + threadIdx.x;     // 64 blocks x 256
    int b  = t >> 10;
    int hw = t & 1023;
    const float* p = z + (size_t)b * (DIM * 1024) + hw;
    float acc = 0.0f;
    #pragma unroll 8
    for (int c = 0; c < DIM; ++c) {
        float v = p[c * 1024];
        acc += v * v;
    }
    float nrm = sqrtf(acc);
    __shared__ float red[256];
    red[threadIdx.x] = nrm;
    __syncthreads();
    for (int s = 128; s > 0; s >>= 1) {
        if (threadIdx.x < s) red[threadIdx.x] += red[threadIdx.x + s];
        __syncthreads();
    }
    if (threadIdx.x == 0) atomicAdd(&sums[0], red[0]);
}

// ---------------------------------------------------------------------------
// 2) scale = (pre_len >= 30) ? 30/pre_len : 1
// ---------------------------------------------------------------------------
__global__ void k_scale(float* __restrict__ sums) {
    float pre = sums[0] * (1.0f / (float)NTOK);
    sums[1] = (pre >= FIXLEN) ? (FIXLEN / pre) : 1.0f;
}

// ---------------------------------------------------------------------------
// 3) embedding -> bf16 (row-major [n][256]) + e2[n] = |e_n|^2 (fp32)
// ---------------------------------------------------------------------------
__global__ void k_emb_bf16(const float* __restrict__ emb,
                           unsigned short* __restrict__ eb,
                           float* __restrict__ e2) {
    int row = blockIdx.x;                 // 16384 blocks x 256 threads
    int c   = threadIdx.x;
    float v = emb[(size_t)row * DIM + c];
    eb[(size_t)row * DIM + c] = f2bf(v);
    __shared__ float red[256];
    red[c] = v * v;
    __syncthreads();
    for (int s = 128; s > 0; s >>= 1) {
        if (c < s) red[c] += red[c + s];
        __syncthreads();
    }
    if (c == 0) e2[row] = red[0];
}

// ---------------------------------------------------------------------------
// 4) zs[t][c] = z[b][c][hw] * scale / 30, bf16, token-major. LDS transpose.
// ---------------------------------------------------------------------------
__global__ void k_zs(const float* __restrict__ z, const float* __restrict__ sums,
                     unsigned short* __restrict__ zs) {
    __shared__ float tile[32][33];
    int t0 = blockIdx.x * 32;             // 1024 % 32 == 0: no b straddle
    int c0 = blockIdx.y * 32;
    int b  = t0 >> 10;
    int hw0 = t0 & 1023;
    float s = sums[1] * (1.0f / FIXLEN);
    #pragma unroll
    for (int i = 0; i < 4; ++i) {
        int cl = threadIdx.y + i * 8;
        tile[cl][threadIdx.x] =
            z[(size_t)b * (DIM * 1024) + (size_t)(c0 + cl) * 1024 + hw0 + threadIdx.x];
    }
    __syncthreads();
    #pragma unroll
    for (int i = 0; i < 4; ++i) {
        int tl = threadIdx.y + i * 8;
        float v = tile[threadIdx.x][tl] * s;
        zs[(size_t)(t0 + tl) * DIM + c0 + threadIdx.x] = f2bf(v);
    }
}

// ---------------------------------------------------------------------------
// async stage copy: 32 embedding rows (16 KB) -> LDS, rows padded 512B->528B.
// 256 threads x 4 global_load_async_to_lds_b128 (ASYNCcnt-tracked).
// ---------------------------------------------------------------------------
__device__ __forceinline__ void stage_copy(const unsigned short* __restrict__ eb,
                                           int stage, unsigned short* lbuf, int tid) {
    const char* gbase = (const char*)eb + (size_t)stage * (STAGE_ROWS * 512);
    unsigned int lbase = (unsigned int)(unsigned long long)lbuf;
    #pragma unroll
    for (int r = 0; r < 4; ++r) {
        int g   = tid + r * 256;          // 16B chunk id, 1024 chunks per stage
        int row = g >> 5;                 // 32 chunks per 512B row
        int col = g & 31;
        unsigned int loff = lbase + (unsigned int)(row * (ROW_US * 2) + col * 16);
        const char* ga = gbase + (size_t)g * 16;
        asm volatile("global_load_async_to_lds_b128 %0, %1, off"
                     :: "v"(loff), "v"(ga) : "memory");
    }
}

// ---------------------------------------------------------------------------
// 5) WMMA distance GEMM + argmin, B staged through double-buffered LDS.
//    8 waves/block share each B stage (8x less global B traffic); async copy
//    of stage s+1 overlaps the 16 WMMAs of stage s.
// ---------------------------------------------------------------------------
__global__ void __launch_bounds__(256) k_argmin(const unsigned short* __restrict__ zs,
                                                const unsigned short* __restrict__ eb,
                                                const float* __restrict__ e2,
                                                int* __restrict__ idxOut) {
    __shared__ alignas(16) unsigned short smem[2][STAGE_ROWS * ROW_US]; // 2 x 16.5 KB

    const int tid   = threadIdx.x;
    const int lane  = tid & 31;
    const int wave  = tid >> 5;
    const int mtile = blockIdx.x * 8 + wave;      // 128 blocks * 8 waves = 1024 tiles
    const int l15   = lane & 15;
    const int hi    = lane >> 4;                  // 0: low K half, 1: high K half

    // --- A tile: 16x32 bf16, contiguous K order (k = 16*hi + s) per K-chunk, pinned in 64 VGPRs ---
    v16bf A[8];
    const unsigned short* arow = zs + (size_t)(mtile * 16 + l15) * DIM + hi * 16;
    #pragma unroll
    for (int kc = 0; kc < 8; ++kc) {
        const unsigned short* p = arow + kc * 32;
        V16U un;
        un.u[0] = *(const uint4*)(p);        // K + 0..7
        un.u[1] = *(const uint4*)(p + 8);    // K + 8..15
        A[kc] = un.v;
    }

    float best[8];
    int   bidx[8];
    #pragma unroll
    for (int r = 0; r < 8; ++r) { best[r] = 3.402823e38f; bidx[r] = 0; }

    // prologue: stage 0
    stage_copy(eb, 0, smem[0], tid);
    asm volatile("s_wait_asynccnt 0x0" ::: "memory");
    __syncthreads();

    for (int s = 0; s < NSTAGES; ++s) {
        // issue next stage into the other buffer (wraps on last iter: harmless)
        stage_copy(eb, (s + 1) & (NSTAGES - 1), smem[(s + 1) & 1], tid);

        const unsigned short* lb = smem[s & 1];
        #pragma unroll
        for (int sub = 0; sub < 2; ++sub) {
            int nb = s * STAGE_ROWS + sub * 16;
            // B 32x16 layout: lane n=l15 holds code row, K = kc*32 + hi*16 + {0..15}
            const unsigned short* brow = lb + (sub * 16 + l15) * ROW_US + hi * 16;
            // load full B tile first: breaks WAR serialization between WMMAs
            v16bf B[8];
            #pragma unroll
            for (int kc = 0; kc < 8; ++kc) {
                const unsigned short* p = brow + kc * 32;
                V16U un;
                un.u[0] = *(const uint4*)(p);
                un.u[1] = *(const uint4*)(p + 8);
                B[kc] = un.v;
            }
            v8f acc = {};
            #pragma unroll
            for (int kc = 0; kc < 8; ++kc)
                acc = __builtin_amdgcn_wmma_f32_16x16x32_bf16(
                    false, A[kc], false, B[kc], (short)0, acc, false, false);

            float e2n = e2[nb + l15];
            int   n   = nb + l15;
            #pragma unroll
            for (int r = 0; r < 8; ++r) {
                float sc = e2n - 2.0f * acc[r];   // |zs|^2 constant per row: dropped
                if (sc < best[r]) { best[r] = sc; bidx[r] = n; }  // monotonic nb => first-min
            }
        }
        // next stage's async copies done + all waves done reading this buffer
        asm volatile("s_wait_asynccnt 0x0" ::: "memory");
        __syncthreads();
    }

    // butterfly min-reduce across the 16 lanes of each half (N columns)
    #pragma unroll
    for (int off = 1; off < 16; off <<= 1) {
        #pragma unroll
        for (int r = 0; r < 8; ++r) {
            float ob = __shfl_xor(best[r], off, 32);
            int   oi = __shfl_xor(bidx[r], off, 32);
            if (ob < best[r] || (ob == best[r] && oi < bidx[r])) {
                best[r] = ob; bidx[r] = oi;
            }
        }
    }
    // C layout: lanes 0-15 -> rows 0..7 (VGPR r), lanes 16-31 -> rows 8..15
    if (l15 == 0) {
        #pragma unroll
        for (int r = 0; r < 8; ++r)
            idxOut[mtile * 16 + hi * 8 + r] = bidx[r];
    }
}

// ---------------------------------------------------------------------------
// 6) z_q = emb[idx]*30 (NCHW), accumulate sum((z_q - zt)^2)
// ---------------------------------------------------------------------------
__global__ void k_out(const float* __restrict__ z, const float* __restrict__ emb,
                      const int* __restrict__ idx, const float* __restrict__ sums,
                      float* __restrict__ out, float* __restrict__ diffsum) {
    int g  = blockIdx.x * 256 + threadIdx.x;   // 16384 blocks
    int hw = g & 1023;
    int c  = (g >> 10) & 255;
    int b  = g >> 18;
    int t  = b * 1024 + hw;
    float zq = emb[(size_t)idx[t] * DIM + c] * FIXLEN;
    float zt = z[g] * sums[1];
    out[g] = zq;                                // straight-through == z_q numerically
    float d = zq - zt; d *= d;
    __shared__ float red[256];
    red[threadIdx.x] = d;
    __syncthreads();
    for (int s = 128; s > 0; s >>= 1) {
        if (threadIdx.x < s) red[threadIdx.x] += red[threadIdx.x + s];
        __syncthreads();
    }
    if (threadIdx.x == 0) atomicAdd(diffsum, red[0]);
}

// ---------------------------------------------------------------------------
// 7) finalize: diff scalar + idx as float tail
// ---------------------------------------------------------------------------
__global__ void k_final(const int* __restrict__ idx, const float* __restrict__ diffsum,
                        float* __restrict__ out) {
    int t = blockIdx.x * 256 + threadIdx.x;    // 64 blocks
    out[ZQ_ELEMS + 1 + t] = (float)idx[t];
    if (t == 0) out[ZQ_ELEMS] = 0.25f * (*diffsum) * (1.0f / (float)ZQ_ELEMS);
}

// ---------------------------------------------------------------------------
extern "C" void kernel_launch(void* const* d_in, const int* in_sizes, int n_in,
                              void* d_out, int out_size, void* d_ws, size_t ws_size,
                              hipStream_t stream) {
    const float* z   = (const float*)d_in[0];   // [16,256,32,32]
    const float* emb = (const float*)d_in[1];   // [16384,256]
    float*       out = (float*)d_out;

    char* ws = (char*)d_ws;
    float*          sums = (float*)ws;                         // [0]=norm_sum [1]=scale [2]=diffsum
    unsigned short* zs   = (unsigned short*)(ws + 256);                       // 8 MB
    unsigned short* eb   = (unsigned short*)(ws + 256 + 8388608);             // 8 MB
    float*          e2   = (float*)(ws + 256 + 16777216);                     // 64 KB
    int*            idx  = (int*)  (ws + 256 + 16777216 + 65536);             // 64 KB

    hipMemsetAsync(sums, 0, 16, stream);

    k_norm    <<<NTOK / 256, 256, 0, stream>>>(z, sums);
    k_scale   <<<1, 1, 0, stream>>>(sums);
    k_emb_bf16<<<NEMB, DIM, 0, stream>>>(emb, eb, e2);
    k_zs      <<<dim3(NTOK / 32, DIM / 32), dim3(32, 8), 0, stream>>>(z, sums, zs);
    k_argmin  <<<NTOK / 16 / 8, 256, 0, stream>>>(zs, eb, e2, idx);
    k_out     <<<ZQ_ELEMS / 256, 256, 0, stream>>>(z, emb, idx, sums, out, &sums[2]);
    k_final   <<<NTOK / 256, 256, 0, stream>>>(idx, &sums[2], out);
}
